// SceneLocalization_90563680403472
// MI455X (gfx1250) — matrix operand order repaired and checked by reading, not compiled, not run on hardware
//
#include <hip/hip_runtime.h>
#include <hip/hip_bf16.h>

// ---------- vector types ----------
typedef __attribute__((ext_vector_type(16))) _Float16 v16h;
typedef __attribute__((ext_vector_type(8)))  _Float16 v8h;
typedef __attribute__((ext_vector_type(4)))  _Float16 f16x4;
typedef __attribute__((ext_vector_type(2)))  _Float16 f16x2;
typedef __attribute__((ext_vector_type(8)))  float    v8f;
typedef __attribute__((ext_vector_type(4)))  float    f32x4;

template <typename T> struct VecOf;
template <> struct VecOf<float>    { using type = f32x4; };
template <> struct VecOf<_Float16> { using type = f16x4; };

// ---------- tiled WMMA GEMM: C = relu(A @ B + bias) ----------
// A: [M,K] (f32 or f16, row-major)   B: [K,N] f32 row-major
// C: [M,N] (f32 or f16)
// Block tile 128x128, BK=32, 256 threads (8 wave32s in 2x4), double-buffered LDS.
constexpr int BM = 128;
constexpr int BN = 128;
constexpr int BK = 32;
constexpr int SA = BK + 8;   // 40 halves = 80B row stride: 16B-aligned, bank-spread
constexpr int SB = BK + 8;

template <typename AT, typename OT>
__global__ __launch_bounds__(256) void gemm_bias_relu(
    const AT* __restrict__ A, const float* __restrict__ Bw,
    const float* __restrict__ bias, OT* __restrict__ C,
    int M, int N, int K)
{
    using AV4 = typename VecOf<AT>::type;

    __shared__ __align__(16) _Float16 As[2][BM * SA];
    __shared__ __align__(16) _Float16 Bs[2][BN * SB];

    const int tid = threadIdx.x;
    const int bm  = blockIdx.y * BM;
    const int bn  = blockIdx.x * BN;

    // A staging: 8 x vec4 per 32-elem K-row, 32 rows/pass, 4 passes
    const int a_k4 = tid & 7;
    const int a_m  = tid >> 3;
    // B staging: each thread owns a 2-row x 8-col f32 patch of the 32x128 tile
    const int b_n8 = tid & 15;   // 16 groups of 8 columns
    const int b_kp = tid >> 4;   // 16 k-pairs -> rows 2*b_kp, 2*b_kp+1

    const int KT = K / BK;

    auto load_stage = [&](int kt, AV4 (&ra)[4], f32x4 (&rb)[4]) {
#pragma unroll
        for (int p = 0; p < 4; ++p)
            ra[p] = *(const AV4*)&A[(size_t)(bm + a_m + p * 32) * K + kt * BK + a_k4 * 4];
#pragma unroll
        for (int r = 0; r < 2; ++r)
#pragma unroll
            for (int c = 0; c < 2; ++c)
                rb[r * 2 + c] = *(const f32x4*)
                    &Bw[(size_t)(kt * BK + 2 * b_kp + r) * N + bn + b_n8 * 8 + c * 4];
    };
    auto store_stage = [&](int buf, AV4 (&ra)[4], f32x4 (&rb)[4]) {
#pragma unroll
        for (int p = 0; p < 4; ++p) {
            f16x4 h;
            h.x = (_Float16)ra[p].x; h.y = (_Float16)ra[p].y;
            h.z = (_Float16)ra[p].z; h.w = (_Float16)ra[p].w;
            *(f16x4*)&As[buf][(a_m + p * 32) * SA + a_k4 * 4] = h;
        }
        // transposed B tile [n][k]: one packed k-pair (b32) per column
#pragma unroll
        for (int j = 0; j < 8; ++j) {
            const float x0 = (j < 4) ? rb[0][j] : rb[1][j - 4];   // row 2*b_kp
            const float x1 = (j < 4) ? rb[2][j] : rb[3][j - 4];   // row 2*b_kp+1
            f16x2 p;
            p.x = (_Float16)x0;
            p.y = (_Float16)x1;
            *(f16x2*)&Bs[buf][(b_n8 * 8 + j) * SB + 2 * b_kp] = p;
        }
    };

    const int wave = tid >> 5;       // wave32
    const int lane = tid & 31;
    const int wm   = wave >> 2;      // 0..1  -> 64 rows each
    const int wn   = wave & 3;       // 0..3  -> 32 cols each
    const int g    = lane >> 4;      // lane group (0/1)
    const int l16  = lane & 15;

    v8f acc[4][2] = {};

    AV4   ra[4];
    f32x4 rb[4];
    load_stage(0, ra, rb);
    store_stage(0, ra, rb);
    __syncthreads();

    for (int kt = 0; kt < KT; ++kt) {
        const int  cur      = kt & 1;
        const bool has_next = (kt + 1) < KT;
        if (has_next) load_stage(kt + 1, ra, rb);

        // A fragments: 16-bit A 16x32 layout — chunks at k = g*8 and k = 16+g*8
        v16h af[4];
#pragma unroll
        for (int t = 0; t < 4; ++t) {
            const _Float16* base = &As[cur][(wm * 64 + t * 16 + l16) * SA];
            v8h lo = *(const v8h*)(base + g * 8);
            v8h hi = *(const v8h*)(base + 16 + g * 8);
            af[t] = __builtin_shufflevector(lo, hi, 0,1,2,3,4,5,6,7,8,9,10,11,12,13,14,15);
        }
        // B fragments: 32x16 layout — lane n = l16, chunks at k = g*8 and k = 16+g*8
        v16h bf[2];
#pragma unroll
        for (int t = 0; t < 2; ++t) {
            const _Float16* base = &Bs[cur][(wn * 32 + t * 16 + l16) * SB];
            v8h lo = *(const v8h*)(base + g * 8);
            v8h hi = *(const v8h*)(base + 16 + g * 8);
            bf[t] = __builtin_shufflevector(lo, hi, 0,1,2,3,4,5,6,7,8,9,10,11,12,13,14,15);
        }
#pragma unroll
        for (int i = 0; i < 4; ++i)
#pragma unroll
            for (int j = 0; j < 2; ++j)
                acc[i][j] = __builtin_amdgcn_wmma_f32_16x16x32_f16(
                    false, af[i], false, bf[j], (short)0, acc[i][j], false, false);

        if (has_next) store_stage(cur ^ 1, ra, rb);
        __syncthreads();
    }

    // epilogue: bias + ReLU.  C/D layout: VGPR r -> m = r + 8*g, n = l16
#pragma unroll
    for (int i = 0; i < 4; ++i) {
        const int rowb = bm + wm * 64 + i * 16 + g * 8;
#pragma unroll
        for (int j = 0; j < 2; ++j) {
            const int   col = bn + wn * 32 + j * 16 + l16;
            const float bv  = bias[col];
#pragma unroll
            for (int r = 0; r < 8; ++r) {
                float v = acc[i][j][r] + bv;
                v = v > 0.f ? v : 0.f;
                C[(size_t)(rowb + r) * N + col] = (OT)v;
            }
        }
    }
}

// ---------- routed per-scene head: wave-per-row ----------
__global__ __launch_bounds__(256) void route_kernel(
    const float* __restrict__ shr, const int* __restrict__ idx,
    const float* __restrict__ Ws, const float* __restrict__ bs,
    float* __restrict__ out, int B)
{
    const int row  = blockIdx.x * 8 + (threadIdx.x >> 5);
    const int lane = threadIdx.x & 31;
    if (row >= B) return;

    const float* srow = shr + (size_t)row * 256;
    const int    s    = idx[row];
    const float* w    = Ws + (size_t)s * 256 * 2;

    float a0 = 0.f, a1 = 0.f;
#pragma unroll
    for (int e = lane; e < 256; e += 32) {
        const float x = srow[e];
        a0 += x * w[e * 2 + 0];
        a1 += x * w[e * 2 + 1];
    }
#pragma unroll
    for (int off = 16; off > 0; off >>= 1) {
        a0 += __shfl_down(a0, off);
        a1 += __shfl_down(a1, off);
    }
    if (lane == 0) {
        out[row * 2 + 0] = a0 + bs[s * 2 + 0];
        out[row * 2 + 1] = a1 + bs[s * 2 + 1];
    }
}

extern "C" void kernel_launch(void* const* d_in, const int* in_sizes, int n_in,
                              void* d_out, int out_size, void* d_ws, size_t ws_size,
                              hipStream_t stream) {
    const float* feat      = (const float*)d_in[0];   // [16384, 7680]
    const int*   scene_idx = (const int*)  d_in[1];   // [16384]
    const float* W1        = (const float*)d_in[2];   // [7680, 512]
    const float* b1        = (const float*)d_in[3];   // [512]
    const float* W2        = (const float*)d_in[4];   // [512, 256]
    const float* b2        = (const float*)d_in[5];   // [256]
    const float* Ws        = (const float*)d_in[6];   // [72, 256, 2]
    const float* bs        = (const float*)d_in[7];   // [72, 2]
    float*       out       = (float*)d_out;           // [16384, 2]

    constexpr int Bn = 16384, F = 7680, H = 512, H2 = 256;

    _Float16* C1 = (_Float16*)d_ws;                                   // [B,H] f16
    float*    C2 = (float*)((char*)d_ws + (size_t)Bn * H * sizeof(_Float16)); // [B,H2] f32

    dim3 blk(256);
    // stage 1: visual = relu(feat @ W1 + b1)  -> f16
    gemm_bias_relu<float, _Float16><<<dim3(H / BN, Bn / BM), blk, 0, stream>>>(
        feat, W1, b1, C1, Bn, H, F);
    // stage 2: shared = relu(visual @ W2 + b2) -> f32
    gemm_bias_relu<_Float16, float><<<dim3(H2 / BN, Bn / BM), blk, 0, stream>>>(
        C1, W2, b2, C2, Bn, H2, H);
    // stage 3: routed per-scene linear
    route_kernel<<<dim3(Bn / 8), blk, 0, stream>>>(C2, scene_idx, Ws, bs, out, Bn);
}
